// Model_12429635355006
// MI455X (gfx1250) — matrix operand order repaired and checked by reading, not compiled, not run on hardware
//
#include <hip/hip_runtime.h>
#include <hip/hip_bf16.h>
#include <stdint.h>

#define VOCAB 50000
#define EMB   256
#define HID   512
#define BATCH 128
#define SEQ   512

typedef __attribute__((ext_vector_type(16))) __bf16 v16bf;
typedef __attribute__((ext_vector_type(8)))  float  v8f;
typedef __attribute__((ext_vector_type(4)))  int    v4i;

#define AS1 __attribute__((address_space(1)))
#define AS3 __attribute__((address_space(3)))

#if __has_builtin(__builtin_amdgcn_global_load_async_to_lds_b128) && \
    __has_builtin(__builtin_amdgcn_s_wait_asynccnt)
#define HAS_ASYNC_LDS 1
#else
#define HAS_ASYNC_LDS 0
#endif

__device__ __forceinline__ unsigned short f2bf(float f){
  union { float f; uint32_t u; } cv; cv.f = f;
  uint32_t u = cv.u;
  u += 0x7FFFu + ((u >> 16) & 1u);          // round-to-nearest-even
  return (unsigned short)(u >> 16);
}
__device__ __forceinline__ float bf2f(unsigned short h){
  union { uint32_t u; float f; } cv; cv.u = ((uint32_t)h) << 16;
  return cv.f;
}
// A-fragment (16x32 bf16) position for k-within-tile kk: lane = halfA*16 + m, elem j
__device__ __forceinline__ void a_pos(int kk, int& halfA, int& j){
  halfA = kk >> 4; j = kk & 15;
}

// Stage 32 bytes global -> LDS. Async (ASYNCcnt-tracked) when available.
__device__ __forceinline__ void stage32(const void* g, void* l){
#if HAS_ASYNC_LDS
  AS1 v4i* gp = (AS1 v4i*)g;                 // b128 async copy, typed per builtin sig
  AS3 v4i* lp = (AS3 v4i*)l;
  __builtin_amdgcn_global_load_async_to_lds_b128(gp, lp, 0, 0);
  __builtin_amdgcn_global_load_async_to_lds_b128(gp, lp, 16, 0);
#else
  ((uint4*)l)[0] = ((const uint4*)g)[0];
  ((uint4*)l)[1] = ((const uint4*)g)[1];
#endif
}
__device__ __forceinline__ void stage_wait(){
#if HAS_ASYNC_LDS
  __builtin_amdgcn_s_wait_asynccnt(0);
#endif
}

// ---------------------------------------------------------------------------
// Kernel 0: pack W (f32, [EMB+HID, HID]) into bf16 WMMA B-fragment layout.
// B (32x16 bf16): lane = halfB*16 + n_local holds k = kt*32 + halfB*16 + j.
// Tile(nt,kt) = 32 lanes x 16 bf16 contiguous (1KB) -> b128-pair loads.
// ---------------------------------------------------------------------------
__global__ void pack_weights(const float* __restrict__ W,
                             unsigned short* __restrict__ pWx,
                             unsigned short* __restrict__ pWh){
  int i = blockIdx.x * blockDim.x + threadIdx.x;
  const int PWX = EMB * HID;     // 131072
  const int PWH = HID * HID;     // 262144
  if (i >= PWX + PWH) return;
  unsigned short* dst; int idx, numKt, baseRow;
  if (i < PWX){ dst = pWx; idx = i;        numKt = EMB/32; baseRow = 0;   }
  else        { dst = pWh; idx = i - PWX;  numKt = HID/32; baseRow = EMB; }
  int tile   = idx >> 9;          // 512 elems per tile
  int within = idx & 511;
  int lane   = within >> 4;
  int j      = within & 15;
  int nt = tile / numKt;
  int kt = tile - nt * numKt;
  int n  = nt * 16 + (lane & 15);
  int k  = kt * 32 + ((lane >> 4) << 4) + j;       // halfB*16 + j
  dst[idx] = f2bf(W[(size_t)(baseRow + k) * HID + n]);
}

// ---------------------------------------------------------------------------
// Kernel 1: xs[b,s,:] = bf16( V[tokens[b,s]] @ Wx )  — gathered WMMA GEMM.
// One workgroup = one M-tile (16 flattened (b,s) rows) x all 512 cols.
// 8 waves, 4 N-tiles per wave; K-loop = 8 tiles of 32 over EMB.
// ---------------------------------------------------------------------------
__global__ void __launch_bounds__(256)
embed_project(const int* __restrict__ tokens, const float* __restrict__ V,
              const unsigned short* __restrict__ pWx,
              unsigned short* __restrict__ xs){
  __shared__ __attribute__((aligned(32))) unsigned short aTiles[(EMB/32) * 512];
  __shared__ int sTok[16];
  const int tid = threadIdx.x;
  const int Mt  = blockIdx.x;                 // 0..4095
  if (tid < 16) sTok[tid] = tokens[Mt * 16 + tid];
  __syncthreads();
  // Cooperative gather: 16 rows x 256 f32, converted to bf16 A-fragment layout.
  for (int idx = tid; idx < 16 * EMB; idx += 256){
    int row = idx >> 8;                       // e consecutive across tid -> coalesced
    int e   = idx & 255;
    float v = V[(size_t)sTok[row] * EMB + e];
    int kt = e >> 5, kk = e & 31, hA, j;
    a_pos(kk, hA, j);
    aTiles[kt * 512 + (hA * 16 + row) * 16 + j] = f2bf(v);
  }
  __syncthreads();
  const int wave = tid >> 5;
  const int lane = tid & 31;
  const int nl   = lane & 15;
  const int mh   = (lane >> 4) * 8;
  const v16bf* pB = reinterpret_cast<const v16bf*>(pWx);
  for (int q = 0; q < 4; ++q){
    int nt = wave * 4 + q;
    v8f c = {};
    for (int kt = 0; kt < EMB/32; ++kt){
      const v16bf* pA = reinterpret_cast<const v16bf*>(&aTiles[kt * 512]);
      v16bf a = pA[lane];
      v16bf b = pB[(nt * (EMB/32) + kt) * 32 + lane];
      c = __builtin_amdgcn_wmma_f32_16x16x32_bf16(false, a, false, b,
                                                  (short)0, c, false, false);
    }
    #pragma unroll
    for (int r = 0; r < 8; ++r){
      int R = Mt * 16 + mh + r;               // flattened (b,s)
      xs[(size_t)R * HID + nt * 16 + nl] = f2bf(c[r]);
    }
  }
}

// ---------------------------------------------------------------------------
// Kernel 2: sequential scan. 8 workgroups x 16 batch rows, 16 waves each.
// - state (16x512 bf16) double-buffered in LDS, A-fragment layout
// - Wh streamed from L2 (pre-packed B fragments), contiguous b128 loads
// - xs slice for step t+1 staged async (ASYNCcnt) into LDS during step t
// - one s_barrier per step; final y = state@Wd + bd fused via ds_add_f32
// ---------------------------------------------------------------------------
__global__ void __launch_bounds__(512)
rnn_scan(const unsigned short* __restrict__ xs,
         const unsigned short* __restrict__ pWh,
         const float* __restrict__ bias, const float* __restrict__ Wd,
         const float* __restrict__ bd, float* __restrict__ out){
  __shared__ __attribute__((aligned(32))) unsigned short st[2][(HID/32) * 512];   // 2x16KB
  __shared__ __attribute__((aligned(32))) unsigned short xbuf[2][16 * HID];       // 2x16KB
  __shared__ float yAcc[16];
  const int tid  = threadIdx.x;
  const int lane = tid & 31;
  const int wave = tid >> 5;                  // 0..15, owns N-tiles 2w, 2w+1
  const int b0   = blockIdx.x * 16;
  const int nl   = lane & 15;
  const int mh   = (lane >> 4) * 8;
  const int nt0  = wave * 2, nt1 = wave * 2 + 1;
  // staging map: thread -> (row, 32B chunk); one wave stages one 1KB row
  const int srow = tid >> 5;
  const int schk = (tid & 31) * 32;

  for (int i = tid; i < (HID/32) * 512; i += 512) st[0][i] = 0;
  if (tid < 16) yAcc[tid] = 0.f;
  // stage xproj slice for t = 0
  stage32((const char*)xs + ((size_t)(b0 + srow) * SEQ + 0) * (HID * 2) + schk,
          (char*)&xbuf[0][srow * HID] + schk);
  stage_wait();
  __syncthreads();

  const v16bf* pB = reinterpret_cast<const v16bf*>(pWh);
  const float bias0 = bias[nt0 * 16 + nl];
  const float bias1 = bias[nt1 * 16 + nl];
  const float wd0   = Wd[nt0 * 16 + nl];
  const float wd1   = Wd[nt1 * 16 + nl];

  for (int t = 0; t < SEQ; ++t){
    const unsigned short* cur  = st[t & 1];
    unsigned short*       nxt  = st[(t + 1) & 1];
    const unsigned short* xcur = xbuf[t & 1];

    if (t + 1 < SEQ)                          // overlap next step's xproj copy
      stage32((const char*)xs + ((size_t)(b0 + srow) * SEQ + (t + 1)) * (HID * 2) + schk,
              (char*)&xbuf[(t + 1) & 1][srow * HID] + schk);
    if (t + 2 < SEQ)                          // warm L2 one step further ahead
      __builtin_prefetch(&xs[((size_t)(b0 + srow) * SEQ + (t + 2)) * HID], 0, 1);

    v8f c0 = {}, c1 = {};
    #pragma unroll
    for (int r = 0; r < 8; ++r){              // accumulator init: xproj + bias (LDS)
      c0[r] = bf2f(xcur[(mh + r) * HID + nt0 * 16 + nl]) + bias0;
      c1[r] = bf2f(xcur[(mh + r) * HID + nt1 * 16 + nl]) + bias1;
    }

    #pragma unroll 4
    for (int kt = 0; kt < HID/32; ++kt){
      const v16bf* pA = reinterpret_cast<const v16bf*>(&cur[kt * 512]);
      v16bf a  = pA[lane];
      v16bf w0 = pB[(nt0 * (HID/32) + kt) * 32 + lane];
      v16bf w1 = pB[(nt1 * (HID/32) + kt) * 32 + lane];
      c0 = __builtin_amdgcn_wmma_f32_16x16x32_bf16(false, a, false, w0,
                                                   (short)0, c0, false, false);
      c1 = __builtin_amdgcn_wmma_f32_16x16x32_bf16(false, a, false, w1,
                                                   (short)0, c1, false, false);
    }

    #pragma unroll
    for (int r = 0; r < 8; ++r){              // tanh + scatter into next A-frags
      float s0 = tanhf(c0[r]);
      float s1 = tanhf(c1[r]);
      int m = mh + r;
      { int n = nt0 * 16 + nl; int kt2 = n >> 5, kk = n & 31, hA, j; a_pos(kk, hA, j);
        nxt[kt2 * 512 + (hA * 16 + m) * 16 + j] = f2bf(s0); }
      { int n = nt1 * 16 + nl; int kt2 = n >> 5, kk = n & 31, hA, j; a_pos(kk, hA, j);
        nxt[kt2 * 512 + (hA * 16 + m) * 16 + j] = f2bf(s1); }
      if (t == SEQ - 1){
        atomicAdd(&yAcc[m], s0 * wd0);
        atomicAdd(&yAcc[m], s1 * wd1);
      }
    }
    stage_wait();                             // async staging done before barrier
    __syncthreads();
  }
  if (tid < 16) out[b0 + tid] = yAcc[tid] + bd[0];
}

extern "C" void kernel_launch(void* const* d_in, const int* in_sizes, int n_in,
                              void* d_out, int out_size, void* d_ws, size_t ws_size,
                              hipStream_t stream){
  (void)in_sizes; (void)n_in; (void)out_size; (void)ws_size;
  const int*   tokens = (const int*)  d_in[0];
  const float* V      = (const float*)d_in[1];
  const float* W      = (const float*)d_in[2];
  const float* b      = (const float*)d_in[3];
  const float* Wd     = (const float*)d_in[4];
  const float* bd     = (const float*)d_in[5];
  float* out = (float*)d_out;

  char* ws = (char*)d_ws;
  unsigned short* pWx = (unsigned short*)(ws);             // 262144 B
  unsigned short* pWh = (unsigned short*)(ws + 262144);    // 524288 B
  unsigned short* xs  = (unsigned short*)(ws + 786432);    // 67108864 B

  const int packN = EMB * HID + HID * HID;
  pack_weights<<<(packN + 255) / 256, 256, 0, stream>>>(W, pWx, pWh);
  embed_project<<<(BATCH * SEQ) / 16, 256, 0, stream>>>(tokens, V, pWx, xs);
  rnn_scan<<<BATCH / 16, 512, 0, stream>>>(xs, pWh, b, Wd, bd, out);
}
